// TransDeAtt_23613730193708
// MI455X (gfx1250) — matrix operand order repaired and checked by reading, not compiled, not run on hardware
//
#include <hip/hip_runtime.h>

typedef __bf16 bf16;
typedef __attribute__((ext_vector_type(8)))  bf16  v8bf;
typedef __attribute__((ext_vector_type(16))) bf16  v16bf;
typedef __attribute__((ext_vector_type(8)))  float v8f;
typedef __attribute__((ext_vector_type(4)))  unsigned int v4u;
typedef __attribute__((ext_vector_type(8)))  int v8i;
typedef __attribute__((ext_vector_type(4)))  int v4i;

#define N_   8
#define P_   4096        // 64*64
#define SX2_ 256
#define IN1_ 768
#define E1_  256
#define E2_  64
#define EBD_ 128
#define BV_  640
#define A1_  128
#define OT_  256

#if __has_builtin(__builtin_amdgcn_tensor_load_to_lds)
#define HAVE_TDM 1
#else
#define HAVE_TDM 0
#endif

// ---------------------------------------------------------------------------
// TDM: DMA a 2D bf16 tile (tileK elems per row, nRows rows, row stride
// strideElems) from global into LDS at byte offset ldsOff (packed rows).
// Descriptor packing per CDNA5 ISA (groups 0/1; 2D
// tensor -> groups 2/3 zero). 6-arg clang-23 builtin form.
// ---------------------------------------------------------------------------
#if HAVE_TDM
__device__ __forceinline__ void tdm_load_2d(unsigned ldsOff, const bf16* gptr,
                                            int tileK, int nRows,
                                            long strideElems) {
  const unsigned long long ga = (unsigned long long)(size_t)gptr;
  v4u g0;
  g0[0] = 1u;                                   // count=1, user mode, no gather
  g0[1] = ldsOff;                               // lds_addr (bytes)
  g0[2] = (unsigned)(ga & 0xFFFFFFFFu);         // global_addr[31:0]
  g0[3] = (unsigned)((ga >> 32) & 0x1FFFFFFu)   // global_addr[56:32]
          | (2u << 30);                         // type=2 ("image")
  v8i g1;
  g1[0] = (int)(1u << 16);                      // wg_mask=0, data_size=1 (2B)
  g1[1] = (int)(((unsigned)strideElems & 0xFFFFu) << 16);        // tensor_dim0 lo16
  g1[2] = (int)((((unsigned)strideElems >> 16) & 0xFFFFu)        // tensor_dim0 hi16
                | (((unsigned)nRows & 0xFFFFu) << 16));          // tensor_dim1 lo16
  g1[3] = (int)(((unsigned)tileK & 0xFFFFu) << 16);              // tile_dim0
  g1[4] = (int)((unsigned)nRows & 0xFFFFu);                      // tile_dim1 (tile_dim2=0)
  g1[5] = (int)(unsigned)(strideElems & 0xFFFFFFFFu);            // dim0_stride[31:0]
  g1[6] = (int)(unsigned)((strideElems >> 32) & 0xFFFFu);        // dim0_stride[47:32]
  g1[7] = 0;
  const v4i gz4 = {0, 0, 0, 0};
  const v8i gz8 = {0, 0, 0, 0, 0, 0, 0, 0};
  __builtin_amdgcn_tensor_load_to_lds(g0, g1, gz4, gz4, gz8, 0);
}
#endif

// ---------------------------------------------------------------------------
// WMMA fragment compute for one K-step out of the given LDS buffers.
// ldsW: [64][32] bf16 (M x K), ldsX: [128][32] bf16 (pixel x K, transposed).
// ---------------------------------------------------------------------------
__device__ __forceinline__ void mma_step(const bf16* lW, const bf16* lX,
                                         v8f (&c)[2][2]) {
  const int lane = threadIdx.x & 31;
  const int wave = threadIdx.x >> 5;
  const int wm = wave & 1, wn = wave >> 1, ml = lane & 15;
  const int kb = (lane & 16) ? 16 : 0, kb2 = (lane & 16) ? 16 : 0;
#pragma unroll
  for (int i = 0; i < 2; ++i) {
    v8bf alo = *reinterpret_cast<const v8bf*>(&lW[(wm * 32 + i * 16 + ml) * 32 + kb]);
    v8bf ahi = *reinterpret_cast<const v8bf*>(&lW[(wm * 32 + i * 16 + ml) * 32 + kb + 8]);
    v16bf a = __builtin_shufflevector(alo, ahi, 0, 1, 2, 3, 4, 5, 6, 7, 8, 9, 10,
                                      11, 12, 13, 14, 15);
#pragma unroll
    for (int j = 0; j < 2; ++j) {
      v16bf b = *reinterpret_cast<const v16bf*>(&lX[(wn * 32 + j * 16 + ml) * 32 + kb2]);
      c[i][j] = __builtin_amdgcn_wmma_f32_16x16x32_bf16(false, a, false, b,
                                                        (short)0, c[i][j], false, false);
    }
  }
}

// ---------------------------------------------------------------------------
// TDM-fed, double-buffered GEMM core. W: [M][K] row-major. X: pixel-major
// [P][ldx]; Xrow0 points at row pBase. Block tile 64(M) x 128(pixels).
// Wave 0 drives the Tensor Data Mover; everyone else just does WMMA.
// ---------------------------------------------------------------------------
__device__ __forceinline__ void mma_core_tdm(const bf16* __restrict__ Wp, int ldw,
                                             const bf16* __restrict__ Xrow0, int ldx,
                                             int Klen, int mBase,
                                             bf16* ldsW, bf16* ldsX, v8f (&c)[2][2]) {
  const int wave = threadIdx.x >> 5;
  const int nIt = Klen >> 5;
#if HAVE_TDM
  const unsigned wOff = (unsigned)(size_t)ldsW;
  const unsigned xOff = (unsigned)(size_t)ldsX;
  if (wave == 0) {
    tdm_load_2d(wOff, Wp + (long)mBase * ldw, 32, 64, ldw);
    tdm_load_2d(xOff, Xrow0, 32, 128, ldx);
  }
  for (int it = 0; it < nIt; ++it) {
    const int cur = it & 1;
    if (wave == 0) __builtin_amdgcn_s_wait_tensorcnt(0);
    __syncthreads();                       // publish buf[cur] to all waves
    if (wave == 0 && it + 1 < nIt) {       // prefetch next tile via TDM
      const int kn = (it + 1) << 5, nb = cur ^ 1;
      tdm_load_2d(wOff + (unsigned)nb * 4096u, Wp + (long)mBase * ldw + kn, 32, 64, ldw);
      tdm_load_2d(xOff + (unsigned)nb * 8192u, Xrow0 + kn, 32, 128, ldx);
    }
    mma_step(ldsW + cur * (64 * 32), ldsX + cur * (128 * 32), c);
    __syncthreads();                       // all reads of buf[cur] done
  }
#else
  // Fallback: per-thread staging (contiguous 16B rows, pixel-major X).
  const int t = threadIdx.x;
  const int rW = t >> 2, cW = (t & 3) << 3;
  const int pX = t & 127, khX = (t >> 7) << 4;
  for (int it = 0; it < nIt; ++it) {
    const int k0 = it << 5;
    __syncthreads();
    *reinterpret_cast<uint4*>(&ldsW[rW * 32 + cW]) =
        *reinterpret_cast<const uint4*>(&Wp[(long)(mBase + rW) * ldw + k0 + cW]);
    *reinterpret_cast<uint4*>(&ldsX[pX * 32 + khX]) =
        *reinterpret_cast<const uint4*>(&Xrow0[(long)pX * ldx + k0 + khX]);
    *reinterpret_cast<uint4*>(&ldsX[pX * 32 + khX + 8]) =
        *reinterpret_cast<const uint4*>(&Xrow0[(long)pX * ldx + k0 + khX + 8]);
    __syncthreads();
    mma_step(ldsW, ldsX, c);
  }
#endif
}

// ---------------------------------------------------------------------------
// Epilogue for pixel-major bf16 outputs: fragments -> per-wave 32x32 LDS tile
// -> each lane stores one contiguous 64B row (4 x b128). Optional elementwise
// multiply by a second pixel-major tensor (same ldx) in the row phase.
// ---------------------------------------------------------------------------
template <class ACT>
__device__ __forceinline__ void epilogue_store_t(v8f (&c)[2][2],
                                                 const float* __restrict__ bias,
                                                 int mBase, int pBase,
                                                 bf16* __restrict__ Out, int ldx,
                                                 const bf16* __restrict__ mulBase,
                                                 bf16* ldsT, ACT act) {
  const int lane = threadIdx.x & 31, wave = threadIdx.x >> 5;
  const int wm = wave & 1, wn = wave >> 1, ml = lane & 15;
  const int hi = (lane & 16) ? 8 : 0;
  bf16* T = ldsT + wave * (32 * 32);
#pragma unroll
  for (int i = 0; i < 2; ++i)
#pragma unroll
    for (int j = 0; j < 2; ++j)
#pragma unroll
      for (int r = 0; r < 8; ++r) {
        const int lm = i * 16 + r + hi;       // local m   (0..31)
        const int lp = j * 16 + ml;           // local pix (0..31)
        const int m = mBase + wm * 32 + lm;
        T[lp * 32 + lm] = (bf16)act(c[i][j][r] + bias[m], m);
      }
  // same-wave LDS ops are in-order; compiler inserts the dscnt waits
  const int gp = pBase + wn * 32 + lane;
  bf16* dst = Out + (long)gp * ldx + mBase + wm * 32;
  if (mulBase) {
    const bf16* mrow = mulBase + (long)gp * ldx + mBase + wm * 32;
#pragma unroll
    for (int q = 0; q < 32; ++q)
      T[lane * 32 + q] = (bf16)((float)T[lane * 32 + q] * (float)mrow[q]);
  }
  const uint4* s = reinterpret_cast<const uint4*>(&T[lane * 32]);
  uint4* d = reinterpret_cast<uint4*>(dst);
#pragma unroll
  for (int q = 0; q < 4; ++q) d[q] = s[q];
}

// ---------------------------------------------------------------------------
// t1: relu(W_t1 @ st + b_t1) -> global-average-pool accumulate only
// ---------------------------------------------------------------------------
__global__ __launch_bounds__(256) void k_t1(const bf16* __restrict__ Wb,
                                            const bf16* __restrict__ stT,
                                            const float* __restrict__ bias,
                                            float* __restrict__ gap) {
  __shared__ alignas(64) bf16 ldsW[2 * 64 * 32];
  __shared__ alignas(64) bf16 ldsX[2 * 128 * 32];
  const int n = blockIdx.z, mBase = blockIdx.y * 64, pBase = blockIdx.x * 128;
  v8f c[2][2] = {};
  mma_core_tdm(Wb, IN1_, stT + ((long)n * P_ + pBase) * IN1_, IN1_, IN1_, mBase,
               ldsW, ldsX, c);
  const int lane = threadIdx.x & 31, wave = threadIdx.x >> 5;
  const int wm = wave & 1, ml = lane & 15, hi = (lane & 16) ? 8 : 0;
#pragma unroll
  for (int i = 0; i < 2; ++i)
#pragma unroll
    for (int j = 0; j < 2; ++j)
#pragma unroll
      for (int r = 0; r < 8; ++r) {
        const int m = mBase + wm * 32 + i * 16 + r + hi;
        float e = fmaxf(c[i][j][r] + bias[m], 0.0f);
        e += __shfl_xor(e, 1, 32);
        e += __shfl_xor(e, 2, 32);
        e += __shfl_xor(e, 4, 32);
        e += __shfl_xor(e, 8, 32);
        if (ml == 0) atomicAdd(&gap[n * E1_ + m], e);
      }
}

// ---------------------------------------------------------------------------
// U: relu(W_U @ sx2 + b_U) * wvec[n][m] -> bU (pixel-major bf16)
// ---------------------------------------------------------------------------
__global__ __launch_bounds__(256) void k_U(const bf16* __restrict__ Wb,
                                           const bf16* __restrict__ XT,
                                           const float* __restrict__ bias,
                                           const float* __restrict__ wvec,
                                           bf16* __restrict__ bU) {
  __shared__ alignas(64) bf16 ldsW[2 * 64 * 32];
  __shared__ alignas(64) bf16 ldsX[2 * 128 * 32];
  __shared__ alignas(64) bf16 ldsT[8 * 32 * 32];
  const int n = blockIdx.z, mBase = blockIdx.y * 64, pBase = blockIdx.x * 128;
  v8f c[2][2] = {};
  mma_core_tdm(Wb, SX2_, XT + ((long)n * P_ + pBase) * SX2_, SX2_, SX2_, mBase,
               ldsW, ldsX, c);
  const float* wv = wvec + n * SX2_;
  epilogue_store_t(c, bias, mBase, pBase, bU + (long)n * P_ * SX2_, SX2_,
                   (const bf16*)nullptr, ldsT,
                   [=](float v, int m) { return fmaxf(v, 0.0f) * wv[m]; });
}

// ---------------------------------------------------------------------------
// V: relu(W_V @ bU + b_V) -> bfull channels [0,128)  (pixel-major)
// ---------------------------------------------------------------------------
__global__ __launch_bounds__(256) void k_V(const bf16* __restrict__ Wb,
                                           const bf16* __restrict__ XT,
                                           const float* __restrict__ bias,
                                           bf16* __restrict__ bfull) {
  __shared__ alignas(64) bf16 ldsW[2 * 64 * 32];
  __shared__ alignas(64) bf16 ldsX[2 * 128 * 32];
  __shared__ alignas(64) bf16 ldsT[8 * 32 * 32];
  const int n = blockIdx.z, mBase = blockIdx.y * 64, pBase = blockIdx.x * 128;
  v8f c[2][2] = {};
  mma_core_tdm(Wb, SX2_, XT + ((long)n * P_ + pBase) * SX2_, SX2_, SX2_, mBase,
               ldsW, ldsX, c);
  epilogue_store_t(c, bias, mBase, pBase, bfull + (long)n * P_ * BV_, BV_,
                   (const bf16*)nullptr, ldsT,
                   [](float v, int) { return fmaxf(v, 0.0f); });
}

// ---------------------------------------------------------------------------
// a1: relu(W_a1 @ bfull + b_a1) -> a1 (pixel-major)
// ---------------------------------------------------------------------------
__global__ __launch_bounds__(256) void k_a1(const bf16* __restrict__ Wb,
                                            const bf16* __restrict__ XT,
                                            const float* __restrict__ bias,
                                            bf16* __restrict__ a1) {
  __shared__ alignas(64) bf16 ldsW[2 * 64 * 32];
  __shared__ alignas(64) bf16 ldsX[2 * 128 * 32];
  __shared__ alignas(64) bf16 ldsT[8 * 32 * 32];
  const int n = blockIdx.z, mBase = blockIdx.y * 64, pBase = blockIdx.x * 128;
  v8f c[2][2] = {};
  mma_core_tdm(Wb, BV_, XT + ((long)n * P_ + pBase) * BV_, BV_, BV_, mBase,
               ldsW, ldsX, c);
  epilogue_store_t(c, bias, mBase, pBase, a1 + (long)n * P_ * A1_, A1_,
                   (const bf16*)nullptr, ldsT,
                   [](float v, int) { return fmaxf(v, 0.0f); });
}

// ---------------------------------------------------------------------------
// a2: sigmoid(W_a2 @ a1 + b_a2) * bfull -> amod (pixel-major)
// ---------------------------------------------------------------------------
__global__ __launch_bounds__(256) void k_a2(const bf16* __restrict__ Wb,
                                            const bf16* __restrict__ XT,
                                            const float* __restrict__ bias,
                                            const bf16* __restrict__ bfull,
                                            bf16* __restrict__ amod) {
  __shared__ alignas(64) bf16 ldsW[2 * 64 * 32];
  __shared__ alignas(64) bf16 ldsX[2 * 128 * 32];
  __shared__ alignas(64) bf16 ldsT[8 * 32 * 32];
  const int n = blockIdx.z, mBase = blockIdx.y * 64, pBase = blockIdx.x * 128;
  v8f c[2][2] = {};
  mma_core_tdm(Wb, A1_, XT + ((long)n * P_ + pBase) * A1_, A1_, A1_, mBase,
               ldsW, ldsX, c);
  epilogue_store_t(c, bias, mBase, pBase, amod + (long)n * P_ * BV_, BV_,
                   bfull + (long)n * P_ * BV_, ldsT,
                   [](float v, int) { return 1.0f / (1.0f + __expf(-v)); });
}

// ---------------------------------------------------------------------------
// conv3x3: implicit GEMM over 9 taps x K=640. amod is pixel-major, so each
// shifted pixel row is a contiguous, predicated 16B load. Output fp32
// channel-major straight into d_out (coalesced across lanes).
// ---------------------------------------------------------------------------
__global__ __launch_bounds__(256) void k_conv3(const bf16* __restrict__ Wotb,
                                               const bf16* __restrict__ amodT,
                                               const float* __restrict__ bias,
                                               float* __restrict__ outA) {
  __shared__ alignas(64) bf16 ldsW[64 * 32];
  __shared__ alignas(64) bf16 ldsX[128 * 32];
  const int t = threadIdx.x;
  const int n = blockIdx.z, mBase = blockIdx.y * 64, pBase = blockIdx.x * 128;
  const bf16* Xp = amodT + (long)n * P_ * BV_;
  const int rW = t >> 2, cW = (t & 3) << 3;
  const int pX = t & 127, khX = (t >> 7) << 4;
  v8f c[2][2] = {};
  for (int tap = 0; tap < 9; ++tap) {
    const int dy = tap / 3 - 1, dx = tap % 3 - 1;
    const int gp = pBase + pX;
    const int y = (gp >> 6) + dy, x = (gp & 63) + dx;
    const bool valid = ((unsigned)y < 64u) && ((unsigned)x < 64u);
    const int sp = valid ? ((y << 6) + x) : 0;
    const bf16* rowp = Xp + (long)sp * BV_;
    const bf16* Wtap = Wotb + (long)tap * OT_ * BV_;
    for (int k0 = 0; k0 < BV_; k0 += 32) {
      __syncthreads();
      *reinterpret_cast<uint4*>(&ldsW[rW * 32 + cW]) =
          *reinterpret_cast<const uint4*>(&Wtap[(long)(mBase + rW) * BV_ + k0 + cW]);
      if (k0 + 32 < BV_)
        __builtin_prefetch(&Wtap[(long)(mBase + rW) * BV_ + k0 + 32 + cW], 0, 0);
      uint4 a0 = *reinterpret_cast<const uint4*>(&rowp[k0 + khX]);
      uint4 a1 = *reinterpret_cast<const uint4*>(&rowp[k0 + khX + 8]);
      if (!valid) { a0 = uint4{0, 0, 0, 0}; a1 = uint4{0, 0, 0, 0}; }
      *reinterpret_cast<uint4*>(&ldsX[pX * 32 + khX]) = a0;
      *reinterpret_cast<uint4*>(&ldsX[pX * 32 + khX + 8]) = a1;
      __syncthreads();
      mma_step(ldsW, ldsX, c);
    }
  }
  const int lane = t & 31, wave = t >> 5;
  const int wm = wave & 1, wn = wave >> 1, ml = lane & 15, hi = (lane & 16) ? 8 : 0;
#pragma unroll
  for (int i = 0; i < 2; ++i)
#pragma unroll
    for (int j = 0; j < 2; ++j)
#pragma unroll
      for (int r = 0; r < 8; ++r) {
        const int m = mBase + wm * 32 + i * 16 + r + hi;
        const int po = pBase + wn * 32 + j * 16 + ml;
        outA[((long)n * OT_ + m) * P_ + po] = fmaxf(c[i][j][r] + bias[m], 0.0f);
      }
}

// ---------------------------------------------------------------------------
// prep: fp32->bf16, bilinear 2x upsample (align_corners), concat; all
// activation tensors stored PIXEL-MAJOR [n][p][c] (c contiguous).
// ---------------------------------------------------------------------------
__global__ __launch_bounds__(256) void k_prep(const float* __restrict__ sx1,
                                              const float* __restrict__ sx2,
                                              const float* __restrict__ thx,
                                              const float* __restrict__ tx,
                                              bf16* __restrict__ stT,
                                              bf16* __restrict__ bfullT,
                                              bf16* __restrict__ sx2T) {
  const long seglen = (long)N_ * 256 * P_;
  const long total = 4 * seglen;
  for (long i = (long)blockIdx.x * 256 + threadIdx.x; i < total;
       i += (long)gridDim.x * 256) {
    const int seg = (int)(i / seglen);
    const long q = i % seglen;
    const int cch = (int)(q % 256);
    const int p = (int)((q / 256) % P_);
    const int n = (int)(q / (256L * P_));
    const long row = (long)n * P_ + p;
    if (seg == 0) {
      stT[row * IN1_ + cch] = (bf16)sx1[((long)n * 256 + cch) * P_ + p];
    } else if (seg == 1) {
      const bf16 v = (bf16)thx[((long)n * 256 + cch) * P_ + p];
      stT[row * IN1_ + 256 + cch] = v;
      bfullT[row * BV_ + EBD_ + cch] = v;
    } else if (seg == 2) {
      const int y = p >> 6, x = p & 63;
      const float fy = y * (31.0f / 63.0f), fx = x * (31.0f / 63.0f);
      const int y0 = (int)fy, x0 = (int)fx;
      const float wy = fy - y0, wx = fx - x0;
      const int y1 = min(y0 + 1, 31), x1 = min(x0 + 1, 31);
      const float* base = tx + ((long)n * 256 + cch) * 1024;
      const float v00 = base[y0 * 32 + x0], v01 = base[y0 * 32 + x1];
      const float v10 = base[y1 * 32 + x0], v11 = base[y1 * 32 + x1];
      const float v = (1.0f - wy) * ((1.0f - wx) * v00 + wx * v01) +
                      wy * ((1.0f - wx) * v10 + wx * v11);
      stT[row * IN1_ + 512 + cch] = (bf16)v;
      bfullT[row * BV_ + EBD_ + 256 + cch] = (bf16)v;
    } else {
      sx2T[row * SX2_ + cch] = (bf16)sx2[((long)n * 256 + cch) * P_ + p];
    }
  }
}

// ---------------------------------------------------------------------------
// weight conversion fp32->bf16 (+ W_ot relayout [O][C][9] -> [tap][O][C])
// ---------------------------------------------------------------------------
__global__ __launch_bounds__(256) void k_wconvert(
    const float* __restrict__ Wt1, const float* __restrict__ WU,
    const float* __restrict__ WV, const float* __restrict__ Wa1,
    const float* __restrict__ Wa2, const float* __restrict__ Wot,
    bf16* __restrict__ Wt1b, bf16* __restrict__ WUb, bf16* __restrict__ WVb,
    bf16* __restrict__ Wa1b, bf16* __restrict__ Wa2b, bf16* __restrict__ Wotb) {
  const long s1 = 256L * 768, s2 = 256L * 256, s3 = 128L * 256;
  const long s4 = 128L * 640, s5 = 640L * 128, s6 = 9L * 256 * 640;
  const long total = s1 + s2 + s3 + s4 + s5 + s6;
  for (long i = (long)blockIdx.x * 256 + threadIdx.x; i < total;
       i += (long)gridDim.x * 256) {
    long k = i;
    if (k < s1) { Wt1b[k] = (bf16)Wt1[k]; continue; }
    k -= s1;
    if (k < s2) { WUb[k] = (bf16)WU[k]; continue; }
    k -= s2;
    if (k < s3) { WVb[k] = (bf16)WV[k]; continue; }
    k -= s3;
    if (k < s4) { Wa1b[k] = (bf16)Wa1[k]; continue; }
    k -= s4;
    if (k < s5) { Wa2b[k] = (bf16)Wa2[k]; continue; }
    k -= s5;
    const int tap = (int)(k / (256L * 640));
    const long rem = k % (256L * 640);
    const int m = (int)(rem / 640), cch = (int)(rem % 640);
    Wotb[k] = (bf16)Wot[((long)m * 640 + cch) * 9 + tap];
  }
}

__global__ __launch_bounds__(256) void k_zero(float* __restrict__ gap, int n) {
  const int i = blockIdx.x * 256 + threadIdx.x;
  if (i < n) gap[i] = 0.0f;
}

// ---------------------------------------------------------------------------
// squeeze-excite chain: mean -> relu(W_t2 @ .) -> W_t3 @ . -> L2 normalize
// ---------------------------------------------------------------------------
__global__ __launch_bounds__(256) void k_wstage(const float* __restrict__ gap,
                                                const float* __restrict__ Wt2,
                                                const float* __restrict__ Wt3,
                                                float* __restrict__ wvec,
                                                float* __restrict__ outW) {
  __shared__ float sm[E1_];
  __shared__ float w2[E2_];
  __shared__ float red[256];
  const int n = blockIdx.x, t = threadIdx.x;
  sm[t] = gap[n * E1_ + t] * (1.0f / (float)P_);
  __syncthreads();
  if (t < E2_) {
    float s = 0.0f;
    for (int c2 = 0; c2 < E1_; ++c2) s += Wt2[t * E1_ + c2] * sm[c2];
    w2[t] = fmaxf(s, 0.0f);
  }
  __syncthreads();
  float s = 0.0f;
  for (int c2 = 0; c2 < E2_; ++c2) s += Wt3[t * E2_ + c2] * w2[c2];
  red[t] = s * s;
  __syncthreads();
  for (int off = 128; off > 0; off >>= 1) {
    if (t < off) red[t] += red[t + off];
    __syncthreads();
  }
  const float v = s / fmaxf(sqrtf(red[0]), 1e-12f);
  wvec[n * SX2_ + t] = v;
  outW[n * SX2_ + t] = v;
}

// ---------------------------------------------------------------------------
extern "C" void kernel_launch(void* const* d_in, const int* in_sizes, int n_in,
                              void* d_out, int out_size, void* d_ws,
                              size_t ws_size, hipStream_t stream) {
  const float* sx1 = (const float*)d_in[0];
  const float* sx2 = (const float*)d_in[1];
  const float* thx = (const float*)d_in[2];
  const float* tx  = (const float*)d_in[3];
  const float* Wt1 = (const float*)d_in[4];
  const float* bt1 = (const float*)d_in[5];
  const float* Wt2 = (const float*)d_in[6];
  const float* Wt3 = (const float*)d_in[7];
  const float* WU  = (const float*)d_in[8];
  const float* bUv = (const float*)d_in[9];
  const float* WV  = (const float*)d_in[10];
  const float* bVv = (const float*)d_in[11];
  const float* Wa1 = (const float*)d_in[12];
  const float* ba1 = (const float*)d_in[13];
  const float* Wa2 = (const float*)d_in[14];
  const float* ba2 = (const float*)d_in[15];
  const float* Wot = (const float*)d_in[16];
  const float* bot = (const float*)d_in[17];

  char* ws = (char*)d_ws;
  size_t off = 0;
  auto alloc = [&](size_t bytes) {
    char* p = ws + off;
    off += (bytes + 255) & ~(size_t)255;
    return p;
  };
  bf16* stT    = (bf16*)alloc((size_t)N_ * P_ * IN1_ * 2);
  bf16* sx2T   = (bf16*)alloc((size_t)N_ * P_ * SX2_ * 2);
  bf16* bfullT = (bf16*)alloc((size_t)N_ * P_ * BV_ * 2);
  bf16* bUT    = (bf16*)alloc((size_t)N_ * P_ * SX2_ * 2);
  bf16* a1T    = (bf16*)alloc((size_t)N_ * P_ * A1_ * 2);
  bf16* amodT  = (bf16*)alloc((size_t)N_ * P_ * BV_ * 2);
  float* gap   = (float*)alloc((size_t)N_ * E1_ * 4);
  float* wvec  = (float*)alloc((size_t)N_ * SX2_ * 4);
  bf16* Wt1b   = (bf16*)alloc(256L * 768 * 2);
  bf16* WUb    = (bf16*)alloc(256L * 256 * 2);
  bf16* WVb    = (bf16*)alloc(128L * 256 * 2);
  bf16* Wa1b   = (bf16*)alloc(128L * 640 * 2);
  bf16* Wa2b   = (bf16*)alloc(640L * 128 * 2);
  bf16* Wotb   = (bf16*)alloc(9L * 256 * 640 * 2);

  float* outA = (float*)d_out;
  float* outW = outA + (size_t)N_ * OT_ * P_;

  k_zero<<<dim3((N_ * E1_ + 255) / 256), dim3(256), 0, stream>>>(gap, N_ * E1_);
  k_prep<<<dim3(8192), dim3(256), 0, stream>>>(sx1, sx2, thx, tx, stT, bfullT, sx2T);
  k_wconvert<<<dim3(2048), dim3(256), 0, stream>>>(Wt1, WU, WV, Wa1, Wa2, Wot,
                                                   Wt1b, WUb, WVb, Wa1b, Wa2b, Wotb);
  k_t1<<<dim3(32, 4, N_), dim3(256), 0, stream>>>(Wt1b, stT, bt1, gap);
  k_wstage<<<dim3(N_), dim3(256), 0, stream>>>(gap, Wt2, Wt3, wvec, outW);
  k_U<<<dim3(32, 4, N_), dim3(256), 0, stream>>>(WUb, sx2T, bUv, wvec, bUT);
  k_V<<<dim3(32, 2, N_), dim3(256), 0, stream>>>(WVb, bUT, bVv, bfullT);
  k_a1<<<dim3(32, 2, N_), dim3(256), 0, stream>>>(Wa1b, bfullT, ba1, a1T);
  k_a2<<<dim3(32, 10, N_), dim3(256), 0, stream>>>(Wa2b, a1T, ba2, bfullT, amodT);
  k_conv3<<<dim3(32, 4, N_), dim3(256), 0, stream>>>(Wotb, amodT, bot, outA);
}
